// LSTMModel_4458176053350
// MI455X (gfx1250) — matrix operand order repaired and checked by reading, compiled, not run on hardware
//
#include <hip/hip_runtime.h>
#include <hip/hip_bf16.h>

typedef __attribute__((ext_vector_type(16))) _Float16 v16h;
typedef __attribute__((ext_vector_type(8)))  _Float16 v8h;
typedef __attribute__((ext_vector_type(8)))  float    v8f;

#define BB 4096
#define TT 50
#define II 64
#define HH 32
#define OO 64
#define GG 128   // 4*H
#define SS 24    // steps_to_predict

union HF16 { v16h v; v8h h[2]; };

__device__ __forceinline__ float fsigm(float x) { return 1.0f / (1.0f + __expf(-x)); }
__device__ __forceinline__ float ftanh_(float x) {
  float e = __expf(2.0f * x);
  return (e - 1.0f) / (e + 1.0f);
}

__global__ __launch_bounds__(32) void lstm_wmma_kernel(
    const float* __restrict__ x,   const float* __restrict__ Wih,
    const float* __restrict__ Whh, const float* __restrict__ bih,
    const float* __restrict__ bhh, const float* __restrict__ Wfc,
    const float* __restrict__ bfc, float* __restrict__ out)
{
  // Weights in original row-major layout, converted to f16 (B-fragment = 2x b128)
  __shared__ __align__(16) _Float16 sWih[GG * II];  // [g][k]  16 KB
  __shared__ __align__(16) _Float16 sWhh[GG * HH];  // [g][k]   8 KB
  __shared__ __align__(16) _Float16 sWfc[OO * HH];  // [o][k]   4 KB
  __shared__ __align__(16) _Float16 sIn[16 * II];   // input tile (x or fed-back softmax)
  __shared__ __align__(16) _Float16 sH [16 * HH];   // hidden tile

  const int lane = threadIdx.x;      // one wave per block
  const int n    = lane & 15;        // C/D column within tile
  const int half = lane >> 4;
  const int b0   = blockIdx.x * 16;  // batch row base of this wave's tile

  // ---- stage weights f32 -> f16 into LDS ----
  auto stage = [&](const float* g, _Float16* l, int cnt) {
    for (int p = lane * 4; p < cnt; p += 32 * 4) {
      float4 v = *reinterpret_cast<const float4*>(g + p);
      l[p + 0] = (_Float16)v.x; l[p + 1] = (_Float16)v.y;
      l[p + 2] = (_Float16)v.z; l[p + 3] = (_Float16)v.w;
    }
  };
  stage(Wih, sWih, GG * II);
  stage(Whh, sWhh, GG * HH);
  stage(Wfc, sWfc, OO * HH);
  for (int p = lane; p < 16 * HH; p += 32) sH[p] = (_Float16)0.0f;  // h0 = 0
  __syncthreads();  // single-wave workgroup: degenerates to nop

  // biases folded into accumulator init: per-lane, per N-tile
  float bg[8];
#pragma unroll
  for (int t = 0; t < 8; ++t) bg[t] = bih[t * 16 + n] + bhh[t * 16 + n];
  float bf[4];
#pragma unroll
  for (int t = 0; t < 4; ++t) bf[t] = bfc[t * 16 + n];

  float cs[2][8];  // cell state, D-layout: cs[j][r] = c(m = r+8*half, hcol = j*16+n)
#pragma unroll
  for (int j = 0; j < 2; ++j)
#pragma unroll
    for (int r = 0; r < 8; ++r) cs[j][r] = 0.0f;

  // A fragment (16xK tile, row-major f16 in LDS with row stride `stride`, K-chunk base kc)
  auto loadA = [&](const _Float16* base, int stride, int kc) -> v16h {
    const _Float16* p = base + (lane & 15) * stride + kc + half * 8;
    HF16 a;
    a.h[0] = *reinterpret_cast<const v8h*>(p);        // K = kc + half*8 + {0..7}
    a.h[1] = *reinterpret_cast<const v8h*>(p + 16);   // K = kc + 16 + half*8 + {0..7}
    return a.v;
  };
  // B fragment (KxN=Kx16) from weight LDS stored [N_total][K_total] row-major
  auto loadB = [&](const _Float16* W, int K, int nt, int kc) -> v16h {
    const _Float16* p = W + (nt * 16 + n) * K + kc + half * 8;
    HF16 b;
    b.h[0] = *reinterpret_cast<const v8h*>(p);        // K = kc + half*8 + {0..7}
    b.h[1] = *reinterpret_cast<const v8h*>(p + 16);   // K = kc + 16 + half*8 + {0..7}
    return b.v;
  };

  // stage this timestep's x tile (f32 global -> f16 LDS)
  auto stage_x = [&](int t) {
#pragma unroll
    for (int q = 0; q < 8; ++q) {
      int p4 = lane + 32 * q;   // float4 index within 16x64 tile
      int m  = p4 >> 4;
      int k4 = p4 & 15;
      float4 v = *reinterpret_cast<const float4*>(
          x + ((size_t)(b0 + m) * TT + t) * II + k4 * 4);
      _Float16* d = sIn + m * II + k4 * 4;
      d[0] = (_Float16)v.x; d[1] = (_Float16)v.y;
      d[2] = (_Float16)v.z; d[3] = (_Float16)v.w;
    }
  };

  // one LSTM cell step: gates = sIn@Wih^T + sH@Whh^T + b ; update c, h; write h -> sH
  auto cell = [&]() {
    v16h ax0 = loadA(sIn, II, 0);
    v16h ax1 = loadA(sIn, II, 32);
    v16h ah  = loadA(sH,  HH, 0);
    v8f acc[8];
#pragma unroll
    for (int nt = 0; nt < 8; ++nt) {
#pragma unroll
      for (int r = 0; r < 8; ++r) acc[nt][r] = bg[nt];
      acc[nt] = __builtin_amdgcn_wmma_f32_16x16x32_f16(
          false, ax0, false, loadB(sWih, II, nt, 0),  (short)0, acc[nt], false, false);
      acc[nt] = __builtin_amdgcn_wmma_f32_16x16x32_f16(
          false, ax1, false, loadB(sWih, II, nt, 32), (short)0, acc[nt], false, false);
      acc[nt] = __builtin_amdgcn_wmma_f32_16x16x32_f16(
          false, ah,  false, loadB(sWhh, HH, nt, 0),  (short)0, acc[nt], false, false);
    }
    // gate order (i,f,g,o): tiles {0,1},{2,3},{4,5},{6,7}; elementwise per (lane,reg)
#pragma unroll
    for (int j = 0; j < 2; ++j) {
#pragma unroll
      for (int r = 0; r < 8; ++r) {
        float ig = fsigm (acc[0 + j][r]);
        float fg = fsigm (acc[2 + j][r]);
        float gg = ftanh_(acc[4 + j][r]);
        float og = fsigm (acc[6 + j][r]);
        float cc = fg * cs[j][r] + ig * gg;
        cs[j][r] = cc;
        float hh = og * ftanh_(cc);
        int m = r + 8 * half;
        sH[m * HH + j * 16 + n] = (_Float16)hh;
      }
    }
  };

  // head: softmax(relu(h@Wfc^T + bfc)); store to out[:, s, :]; stage into sIn as next input
  auto head = [&](int s) {
    v16h ah = loadA(sH, HH, 0);
    v8f hacc[4];
#pragma unroll
    for (int nt = 0; nt < 4; ++nt) {
#pragma unroll
      for (int r = 0; r < 8; ++r) hacc[nt][r] = bf[nt];
      hacc[nt] = __builtin_amdgcn_wmma_f32_16x16x32_f16(
          false, ah, false, loadB(sWfc, HH, nt, 0), (short)0, hacc[nt], false, false);
    }
#pragma unroll
    for (int r = 0; r < 8; ++r) {
      float e[4], mx = 0.0f, rs = 0.0f;   // relu floor 0 -> max >= 0
#pragma unroll
      for (int nt = 0; nt < 4; ++nt) {
        float v = fmaxf(hacc[nt][r], 0.0f);
        hacc[nt][r] = v;
        mx = fmaxf(mx, v);
      }
#pragma unroll
      for (int d = 1; d < 16; d <<= 1) mx = fmaxf(mx, __shfl_xor(mx, d, 32));
#pragma unroll
      for (int nt = 0; nt < 4; ++nt) { e[nt] = __expf(hacc[nt][r] - mx); rs += e[nt]; }
#pragma unroll
      for (int d = 1; d < 16; d <<= 1) rs += __shfl_xor(rs, d, 32);
      float inv = 1.0f / rs;
      int m = r + 8 * half;
#pragma unroll
      for (int nt = 0; nt < 4; ++nt) {
        float o = e[nt] * inv;
        out[(size_t)(b0 + m) * (SS * OO) + (size_t)s * OO + nt * 16 + n] = o;
        sIn[m * II + nt * 16 + n] = (_Float16)o;  // autoregressive feedback
      }
    }
  };

  // warmup over the input sequence
  for (int t = 0; t < TT; ++t) { stage_x(t); cell(); }
  head(0);
  // autoregressive steps: input = previous softmax output (already staged in sIn)
  for (int s = 1; s < SS; ++s) { cell(); head(s); }
}

extern "C" void kernel_launch(void* const* d_in, const int* in_sizes, int n_in,
                              void* d_out, int out_size, void* d_ws, size_t ws_size,
                              hipStream_t stream) {
  const float* x    = (const float*)d_in[0];
  const float* Wih  = (const float*)d_in[1];
  const float* Whh  = (const float*)d_in[2];
  const float* bih  = (const float*)d_in[3];
  const float* bhh  = (const float*)d_in[4];
  const float* Wfc  = (const float*)d_in[5];
  const float* bfc  = (const float*)d_in[6];
  float* out        = (float*)d_out;
  (void)in_sizes; (void)n_in; (void)out_size; (void)d_ws; (void)ws_size;

  dim3 grid(BB / 16);   // 256 independent single-wave tiles
  dim3 block(32);
  lstm_wmma_kernel<<<grid, block, 0, stream>>>(x, Wih, Whh, bih, bhh, Wfc, bfc, out);
}
